// A2Attention_23167053594768
// MI455X (gfx1250) — matrix operand order repaired and checked by reading, not compiled, not run on hardware
//
#include <hip/hip_runtime.h>
#include <hip/hip_bf16.h>

// ---------------------------------------------------------------------------
// Problem constants (B=2, S=2048, H=1024, 16 heads, D=64)
// ---------------------------------------------------------------------------
#define BATCH 2
#define SEQ   2048
#define HID   1024
#define NH    16
#define HD    64
#define ROWS  (BATCH * SEQ)   // 4096
#define EPSV  1e-6f

typedef _Float16 f16;
typedef __attribute__((ext_vector_type(16))) _Float16 v16h;
typedef __attribute__((ext_vector_type(8)))  _Float16 v8h;
typedef __attribute__((ext_vector_type(8)))  float    v8f;
typedef __attribute__((ext_vector_type(4)))  unsigned int u32x4;
typedef __attribute__((ext_vector_type(8)))  int          i32x8;
typedef __attribute__((ext_vector_type(4)))  int          i32x4;

#define HAS_TDM __has_builtin(__builtin_amdgcn_tensor_load_to_lds)

static __device__ __forceinline__ v16h mk16(v8h a, v8h b) {
    return __builtin_shufflevector(a, b, 0,1,2,3,4,5,6,7,8,9,10,11,12,13,14,15);
}

// xor-reductions within 16-lane groups (halves of wave32) via ds_swizzle
static __device__ __forceinline__ float redmax16(float x) {
    x = fmaxf(x, __int_as_float(__builtin_amdgcn_ds_swizzle(__float_as_int(x), 0x041F)));
    x = fmaxf(x, __int_as_float(__builtin_amdgcn_ds_swizzle(__float_as_int(x), 0x081F)));
    x = fmaxf(x, __int_as_float(__builtin_amdgcn_ds_swizzle(__float_as_int(x), 0x101F)));
    x = fmaxf(x, __int_as_float(__builtin_amdgcn_ds_swizzle(__float_as_int(x), 0x201F)));
    return x;
}
static __device__ __forceinline__ float redsum16(float x) {
    x += __int_as_float(__builtin_amdgcn_ds_swizzle(__float_as_int(x), 0x041F));
    x += __int_as_float(__builtin_amdgcn_ds_swizzle(__float_as_int(x), 0x081F));
    x += __int_as_float(__builtin_amdgcn_ds_swizzle(__float_as_int(x), 0x101F));
    x += __int_as_float(__builtin_amdgcn_ds_swizzle(__float_as_int(x), 0x201F));
    return x;
}

#if HAS_TDM
// -------------------- Tensor Data Mover (TDM) helpers ----------------------
static __device__ __forceinline__ void setb(unsigned int* a, int lsb, int width,
                                            unsigned long long v) {
    for (int i = 0; i < width; ) {
        int dw = (lsb + i) >> 5, sh = (lsb + i) & 31;
        int n = 32 - sh; if (n > width - i) n = width - i;
        unsigned int m = (n == 32) ? 0xffffffffu : ((1u << n) - 1u);
        a[dw] |= ((unsigned int)(v >> i) & m) << sh;
        i += n;
    }
}

// 2-D tile DMA: global (f16, row stride d0_stride elems) -> LDS (+row padding)
// pad codes: interval: bytes-per-row = 8<<code ; amount: (code+1) DWORDs
static __device__ __forceinline__ void
tdm_load_2d(const void* gbase, unsigned int lds_addr, unsigned long long byte_off,
            unsigned int tensor_d0, unsigned int tensor_d1,
            unsigned long long d0_stride,
            unsigned int tile_d0, unsigned int tile_d1,
            int pad_int_code, int pad_amt_code) {
    unsigned int g0[4] = {0, 0, 0, 0};
    unsigned int g1[8] = {0, 0, 0, 0, 0, 0, 0, 0};
    unsigned long long ga = (unsigned long long)gbase + byte_off;

    setb(g0, 0, 2, 1);                 // count = 1 (valid descriptor)
    g0[1] = lds_addr;                  // bits 63:32 lds_addr
    setb(g0, 64, 57, ga);              // global tile address
    setb(g0, 126, 2, 2);               // type = image

    setb(g1, 16, 2, 1);                // data_size = 2 bytes
    setb(g1, 20, 1, 1);                // pad_enable
    setb(g1, 22, 3, (unsigned)pad_int_code);
    setb(g1, 25, 7, (unsigned)pad_amt_code);
    setb(g1, 48, 32, tensor_d0);       // tensor_dim0
    setb(g1, 80, 32, tensor_d1);       // tensor_dim1
    setb(g1, 112, 16, tile_d0);        // tile_dim0
    setb(g1, 128, 16, tile_d1);        // tile_dim1 (tile_dim2 = 0 -> 2D)
    setb(g1, 160, 48, d0_stride);      // tensor_dim0_stride

    u32x4 v0; i32x8 v1;
    #pragma unroll
    for (int i = 0; i < 4; ++i) v0[i] = g0[i];
    #pragma unroll
    for (int i = 0; i < 8; ++i) v1[i] = (int)g1[i];
    i32x4 z4 = {0, 0, 0, 0};
#if defined(__clang_major__) && (__clang_major__ >= 23)
    i32x8 z8 = {0, 0, 0, 0, 0, 0, 0, 0};
    __builtin_amdgcn_tensor_load_to_lds(v0, v1, z4, z4, z8, 0);
#else
    __builtin_amdgcn_tensor_load_to_lds(v0, v1, z4, z4, 0);
#endif
}
static __device__ __forceinline__ unsigned int lds_off(const void* p) {
    return (unsigned int)(unsigned long long)p;   // low 32 bits = LDS address
}
#endif // HAS_TDM

// ---------------------------------------------------------------------------
// f32 -> f16 convert (4 elems / thread)
// ---------------------------------------------------------------------------
__global__ void k_cvt(const float* __restrict__ x, f16* __restrict__ y) {
    size_t i = ((size_t)blockIdx.x * 256 + threadIdx.x) * 4;
    #pragma unroll
    for (int j = 0; j < 4; ++j) y[i + j] = (f16)x[i + j];
}

// ---------------------------------------------------------------------------
// tiled transpose + convert:  Wt[k*H + n] = (f16) W[n*H + k]
// ---------------------------------------------------------------------------
__global__ void k_trans(const float* __restrict__ W, f16* __restrict__ Wt) {
    __shared__ float t[32][33];
    int n0 = blockIdx.x * 32, k0 = blockIdx.y * 32;
    int tx = threadIdx.x, ty = threadIdx.y;
    #pragma unroll
    for (int i = 0; i < 32; i += 8)
        t[ty + i][tx] = W[(size_t)(n0 + ty + i) * HID + k0 + tx];
    __syncthreads();
    #pragma unroll
    for (int i = 0; i < 32; i += 8)
        Wt[(size_t)(k0 + ty + i) * HID + n0 + tx] = (f16)t[tx][ty + i];
}

// ---------------------------------------------------------------------------
// WMMA GEMM  C[MxN] = A[MxK](f16) * Bt[KxN](f16) + bias
// TDM double-buffered: wave0 DMAs tile t+1 while all waves run WMMAs on t.
// ---------------------------------------------------------------------------
__global__ void __launch_bounds__(256)
k_gemm(const f16* __restrict__ A, const f16* __restrict__ Bt,
       const float* __restrict__ bias,
       float* __restrict__ outF, f16* __restrict__ outH,
       int M, int N, int K) {
    __shared__ f16 As[2][128][40];   // row 80B  (64B data + 16B pad)
    __shared__ f16 Bs[2][32][136];   // row 272B (256B data + 16B pad)

    const int tid  = threadIdx.x;
    const int lane = tid & 31;
    const int wave = tid >> 5;
    const int wm   = wave >> 1;
    const int wn   = wave & 1;
    const int r    = lane & 15;
    const int hh   = lane >> 4;
    const int m0   = blockIdx.y * 128;
    const int n0   = blockIdx.x * 128;

    v8f acc[2][4] = {};
    const int nt = K / 32;

    for (int t = 0; t < nt; ++t) {
        const int cur = t & 1;
#if HAS_TDM
        if (wave == 0) {
            if (t == 0) {   // prime the pipeline
                tdm_load_2d(A,  lds_off(&As[0][0][0]), ((size_t)m0 * K) * 2,
                            K, M, K, 32, 128, /*int*/3, /*amt*/3);
                tdm_load_2d(Bt, lds_off(&Bs[0][0][0]), ((size_t)0 * N + n0) * 2,
                            N, K, N, 128, 32, /*int*/5, /*amt*/3);
            }
            if (t + 1 < nt) {
                int k1 = (t + 1) * 32;
                tdm_load_2d(A,  lds_off(&As[1 - cur][0][0]),
                            ((size_t)m0 * K + k1) * 2, K, M, K, 32, 128, 3, 3);
                tdm_load_2d(Bt, lds_off(&Bs[1 - cur][0][0]),
                            ((size_t)k1 * N + n0) * 2, N, K, N, 128, 32, 5, 3);
                __builtin_amdgcn_s_wait_tensorcnt((short)2); // tile t landed
            } else {
                __builtin_amdgcn_s_wait_tensorcnt((short)0);
            }
        }
#else
        {   // fallback: cooperative synchronous staging
            const int rowA = tid >> 1, partA = tid & 1;
            const f16* gp = A + (size_t)(m0 + rowA) * K + t * 32 + partA * 16;
            *(v8h*)&As[cur][rowA][partA * 16]     = *(const v8h*)gp;
            *(v8h*)&As[cur][rowA][partA * 16 + 8] = *(const v8h*)(gp + 8);
            const int rowB = tid >> 3, partB = tid & 7;
            const f16* gq = Bt + (size_t)(t * 32 + rowB) * N + n0 + partB * 16;
            *(v8h*)&Bs[cur][rowB][partB * 16]     = *(const v8h*)gq;
            *(v8h*)&Bs[cur][rowB][partB * 16 + 8] = *(const v8h*)(gq + 8);
        }
#endif
        __syncthreads();

        v16h a[2], b[4];
        #pragma unroll
        for (int mi = 0; mi < 2; ++mi) {
            const f16* p = &As[cur][wm * 32 + mi * 16 + r][0];
            a[mi] = mk16(*(const v8h*)(p + hh * 8), *(const v8h*)(p + 16 + hh * 8));
        }
        #pragma unroll
        for (int ni = 0; ni < 4; ++ni) {
            const f16* p = &Bs[cur][0][wn * 64 + ni * 16 + r];
            #pragma unroll
            for (int s = 0; s < 16; ++s)
                b[ni][s] = p[((s < 8) ? hh * 8 + s : 16 + hh * 8 + (s - 8)) * 136];
        }
        #pragma unroll
        for (int mi = 0; mi < 2; ++mi)
            #pragma unroll
            for (int ni = 0; ni < 4; ++ni)
                acc[mi][ni] = __builtin_amdgcn_wmma_f32_16x16x32_f16(
                    false, a[mi], false, b[ni], (short)0, acc[mi][ni], false, false);
        __syncthreads();
    }

    #pragma unroll
    for (int mi = 0; mi < 2; ++mi)
        #pragma unroll
        for (int ni = 0; ni < 4; ++ni)
            #pragma unroll
            for (int g = 0; g < 8; ++g) {
                int row = m0 + wm * 32 + mi * 16 + g + hh * 8;
                int col = n0 + wn * 64 + ni * 16 + r;
                float v = acc[mi][ni][g] + bias[col];
                if (outF) outF[(size_t)row * N + col] = v;
                else      outH[(size_t)row * N + col] = (f16)v;
            }
}

// ---------------------------------------------------------------------------
// RMS-norm + RoPE for q,k. Writes Qh [row][H] and Kth [b][h][d][s] (f16).
// ---------------------------------------------------------------------------
__global__ void __launch_bounds__(256)
k_rmsrope(const float* __restrict__ qraw, const float* __restrict__ kraw,
          const float* __restrict__ cosT, const float* __restrict__ sinT,
          const float* __restrict__ w, f16* __restrict__ Qh, f16* __restrict__ Kth) {
    int row = blockIdx.x;
    int b = row >> 11;
    int s = row & (SEQ - 1);
    int tid = threadIdx.x;

    const float* q = qraw + (size_t)row * HID;
    const float* k = kraw + (size_t)row * HID;

    __shared__ float rq[256], rk[256];
    float sq = 0.f, sk = 0.f;
    for (int i = tid; i < HID; i += 256) {
        float a = q[i]; sq += a * a;
        float c = k[i]; sk += c * c;
    }
    rq[tid] = sq; rk[tid] = sk;
    __syncthreads();
    for (int st = 128; st > 0; st >>= 1) {
        if (tid < st) { rq[tid] += rq[tid + st]; rk[tid] += rk[tid + st]; }
        __syncthreads();
    }
    float rsq = rsqrtf(rq[0] * (1.0f / HID) + EPSV);
    float rsk = rsqrtf(rk[0] * (1.0f / HID) + EPSV);

    for (int i = tid; i < HID; i += 256) {
        int d = i & 63, h = i >> 6;
        float cw = cosT[s * HD + d], sw = sinT[s * HD + d];
        int j = (d < 32) ? i + 32 : i - 32;

        float qn = q[i] * rsq * w[i];
        float qp = q[j] * rsq * w[j];
        Qh[(size_t)row * HID + i] = (f16)(qn * cw + ((d < 32) ? -qp : qp) * sw);

        float kn = k[i] * rsk * w[i];
        float kp = k[j] * rsk * w[j];
        Kth[(((size_t)b * NH + h) * HD + d) * SEQ + s] =
            (f16)(kn * cw + ((d < 32) ? -kp : kp) * sw);
    }
}

// ---------------------------------------------------------------------------
// Flash attention with WMMA; Q/K/V tiles staged by TDM.
// Grid (S/128, NH, B), 256 threads; wave owns 16 q-rows.
// ---------------------------------------------------------------------------
__global__ void __launch_bounds__(256)
k_attn(const f16* __restrict__ Qh, const f16* __restrict__ Kth,
       const f16* __restrict__ Vh, f16* __restrict__ Ah) {
    __shared__ f16 Qs[128][72];     // [q][d]
    __shared__ f16 Ks[64][72];      // [d][j]
    __shared__ f16 Vs[64][72];      // [kv][d]
    __shared__ f16 Ps[8][16][72];   // per-wave probability strip

    const int q0   = blockIdx.x * 128;
    const int h    = blockIdx.y;
    const int b    = blockIdx.z;
    const int tid  = threadIdx.x;
    const int lane = tid & 31;
    const int wave = tid >> 5;
    const int r    = lane & 15;
    const int hh   = lane >> 4;

#if HAS_TDM
    if (wave == 0)   // Q tile: 128 rows x 64 f16 (row 128B -> pad code 4/3)
        tdm_load_2d(Qh, lds_off(&Qs[0][0]),
                    ((size_t)(b * SEQ + q0) * HID + h * HD) * 2,
                    HID, ROWS, HID, 64, 128, 4, 3);
#else
    {
        int row = tid >> 1, part = tid & 1;
        const f16* gp = Qh + (size_t)(b * SEQ + q0 + row) * HID + h * HD + part * 32;
        #pragma unroll
        for (int j = 0; j < 4; ++j)
            *(v8h*)&Qs[row][part * 32 + j * 8] = *(const v8h*)(gp + j * 8);
    }
#endif

    v8f O[4] = {};
    float mrow[8], lrow[8];
    #pragma unroll
    for (int g = 0; g < 8; ++g) { mrow[g] = -1e30f; lrow[g] = 0.f; }

    for (int kv0 = 0; kv0 < SEQ; kv0 += 64) {
        __syncthreads();   // previous iteration done with Ks/Vs/Ps
#if HAS_TDM
        if (wave == 0) {
            tdm_load_2d(Kth, lds_off(&Ks[0][0]),
                        (((size_t)(b * NH + h)) * HD * SEQ + kv0) * 2,
                        SEQ, BATCH * NH * HD, SEQ, 64, 64, 4, 3);
            tdm_load_2d(Vh, lds_off(&Vs[0][0]),
                        ((size_t)(b * SEQ + kv0) * HID + h * HD) * 2,
                        HID, ROWS, HID, 64, 64, 4, 3);
            __builtin_amdgcn_s_wait_tensorcnt((short)0);
        }
#else
        {
            int d = tid >> 2, part = tid & 3;
            const f16* gp = Kth + (((size_t)(b * NH + h)) * HD + d) * SEQ + kv0 + part * 16;
            *(v8h*)&Ks[d][part * 16]     = *(const v8h*)gp;
            *(v8h*)&Ks[d][part * 16 + 8] = *(const v8h*)(gp + 8);
            const f16* gq = Vh + (size_t)(b * SEQ + kv0 + d) * HID + h * HD + part * 16;
            *(v8h*)&Vs[d][part * 16]     = *(const v8h*)gq;
            *(v8h*)&Vs[d][part * 16 + 8] = *(const v8h*)(gq + 8);
        }
#endif
        __syncthreads();

        // ---- scores: wave strip [16 q] x [64 kv] ----
        v8f sa[4] = {};
        #pragma unroll
        for (int d0 = 0; d0 < 64; d0 += 32) {
            const f16* pa = &Qs[wave * 16 + r][d0];
            v16h aq = mk16(*(const v8h*)(pa + hh * 8), *(const v8h*)(pa + 16 + hh * 8));
            #pragma unroll
            for (int ni = 0; ni < 4; ++ni) {
                const f16* pb = &Ks[d0][ni * 16 + r];
                v16h bk;
                #pragma unroll
                for (int s = 0; s < 16; ++s)
                    bk[s] = pb[((s < 8) ? hh * 8 + s : 16 + hh * 8 + (s - 8)) * 72];
                sa[ni] = __builtin_amdgcn_wmma_f32_16x16x32_f16(
                    false, aq, false, bk, (short)0, sa[ni], false, false);
            }
        }

        // ---- online softmax (reference adds +1 on lower triangle) ----
        float p[4][8], tmax[8];
        #pragma unroll
        for (int g = 0; g < 8; ++g) tmax[g] = -1e30f;
        #pragma unroll
        for (int ni = 0; ni < 4; ++ni)
            #pragma unroll
            for (int g = 0; g < 8; ++g) {
                int qq = q0 + wave * 16 + g + hh * 8;
                int jj = kv0 + ni * 16 + r;
                float sv = sa[ni][g] * 0.125f + ((jj <= qq) ? 1.0f : 0.0f);
                p[ni][g] = sv;
                tmax[g] = fmaxf(tmax[g], sv);
            }
        #pragma unroll
        for (int g = 0; g < 8; ++g) tmax[g] = redmax16(tmax[g]);

        float corr[8];
        #pragma unroll
        for (int g = 0; g < 8; ++g) {
            float nm = fmaxf(mrow[g], tmax[g]);
            corr[g] = __expf(mrow[g] - nm);
            mrow[g] = nm;
        }
        #pragma unroll
        for (int ni = 0; ni < 4; ++ni)
            #pragma unroll
            for (int g = 0; g < 8; ++g)
                p[ni][g] = __expf(p[ni][g] - mrow[g]);
        #pragma unroll
        for (int g = 0; g < 8; ++g) {
            float srow = 0.f;
            #pragma unroll
            for (int ni = 0; ni < 4; ++ni) srow += p[ni][g];
            srow = redsum16(srow);
            lrow[g] = lrow[g] * corr[g] + srow;
        }
        #pragma unroll
        for (int ni = 0; ni < 4; ++ni)
            #pragma unroll
            for (int g = 0; g < 8; ++g) O[ni][g] *= corr[g];

        // ---- stage P into LDS as A-fragment source ----
        #pragma unroll
        for (int ni = 0; ni < 4; ++ni)
            #pragma unroll
            for (int g = 0; g < 8; ++g)
                Ps[wave][g + hh * 8][ni * 16 + r] = (f16)p[ni][g];
        __syncthreads();

        // ---- O += P * V ----
        #pragma unroll
        for (int k0 = 0; k0 < 64; k0 += 32) {
            const f16* pa = &Ps[wave][r][k0];
            v16h ap = mk16(*(const v8h*)(pa + hh * 8), *(const v8h*)(pa + 16 + hh * 8));
            #pragma unroll
            for (int ni = 0; ni < 4; ++ni) {
                const f16* pb = &Vs[k0][ni * 16 + r];
                v16h bv;
                #pragma unroll
                for (int s = 0; s < 16; ++s)
                    bv[s] = pb[((s < 8) ? hh * 8 + s : 16 + hh * 8 + (s - 8)) * 72];
                O[ni] = __builtin_amdgcn_wmma_f32_16x16x32_f16(
                    false, ap, false, bv, (short)0, O[ni], false, false);
            }
        }
    }

    #pragma unroll
    for (int ni = 0; ni < 4; ++ni)
        #pragma unroll
        for (int g = 0; g < 8; ++g) {
            int row = q0 + wave * 16 + g + hh * 8;
            float ov = O[ni][g] / lrow[g];
            Ah[(size_t)(b * SEQ + row) * HID + h * HD + ni * 16 + r] = (f16)ov;
        }
}

// ---------------------------------------------------------------------------
// Launch
// ---------------------------------------------------------------------------
extern "C" void kernel_launch(void* const* d_in, const int* in_sizes, int n_in,
                              void* d_out, int out_size, void* d_ws, size_t ws_size,
                              hipStream_t stream) {
    const float* x    = (const float*)d_in[0];
    const float* cosT = (const float*)d_in[1];
    const float* sinT = (const float*)d_in[2];
    const float* Wq   = (const float*)d_in[3];
    const float* bq   = (const float*)d_in[4];
    const float* Wk   = (const float*)d_in[5];
    const float* bk   = (const float*)d_in[6];
    const float* Wv   = (const float*)d_in[7];
    const float* bv   = (const float*)d_in[8];
    const float* Wo   = (const float*)d_in[9];
    const float* bo   = (const float*)d_in[10];
    const float* rw   = (const float*)d_in[11];

    char* ws = (char*)d_ws;
    const size_t MB = 1u << 20;
    f16*   Xh   = (f16*)(ws + 0);             // 8 MB
    f16*   Wqt  = (f16*)(ws + 8  * MB);       // 4 x 2 MB
    f16*   Wkt  = Wqt + (size_t)HID * HID;
    f16*   Wvt  = Wkt + (size_t)HID * HID;
    f16*   Wot  = Wvt + (size_t)HID * HID;
    float* qraw = (float*)(ws + 16 * MB);     // 16 MB
    float* kraw = (float*)(ws + 32 * MB);     // 16 MB
    f16*   Qh   = (f16*)(ws + 48 * MB);       // 8 MB
    f16*   Kth  = (f16*)(ws + 56 * MB);       // 8 MB
    f16*   Vh   = (f16*)(ws + 64 * MB);       // 8 MB
    f16*   Ah   = (f16*)(ws + 72 * MB);       // 8 MB

    k_cvt<<<(ROWS * HID) / 1024, 256, 0, stream>>>(x, Xh);
    dim3 tb(32, 8), tg(HID / 32, HID / 32);
    k_trans<<<tg, tb, 0, stream>>>(Wq, Wqt);
    k_trans<<<tg, tb, 0, stream>>>(Wk, Wkt);
    k_trans<<<tg, tb, 0, stream>>>(Wv, Wvt);
    k_trans<<<tg, tb, 0, stream>>>(Wo, Wot);

    dim3 gg(HID / 128, ROWS / 128);
    k_gemm<<<gg, 256, 0, stream>>>(Xh, Wqt, bq, qraw, nullptr, ROWS, HID, HID);
    k_gemm<<<gg, 256, 0, stream>>>(Xh, Wkt, bk, kraw, nullptr, ROWS, HID, HID);
    k_gemm<<<gg, 256, 0, stream>>>(Xh, Wvt, bv, nullptr, Vh,   ROWS, HID, HID);

    k_rmsrope<<<ROWS, 256, 0, stream>>>(qraw, kraw, cosT, sinT, rw, Qh, Kth);

    k_attn<<<dim3(SEQ / 128, NH, BATCH), 256, 0, stream>>>(Qh, Kth, Vh, Ah);

    k_gemm<<<gg, 256, 0, stream>>>(Ah, Wot, bo, (float*)d_out, nullptr, ROWS, HID, HID);
}
